// RackioLSTMCell_88742614270388
// MI455X (gfx1250) — matrix operand order repaired and checked by reading, not compiled, not run on hardware
//
#include <hip/hip_runtime.h>
#include <hip/hip_bf16.h>

// ---------------------------------------------------------------------------
// LSTM scan on MI455X (gfx1250, wave32, WMMA bf16 16x16x32)
//
//   * ALL weights live in VGPRs: w[4][24] fragment array = 768 VGPRs/lane,
//     preloaded once before the scan (1 wave/SIMD -> ideal for back-to-back
//     WMMA issue per ISA 7.5)
//   * c-state in VGPRs; h ping-pongs through a 2x64KB bf16 L2 buffer
//   * A fragments explicitly double-buffered, incl. cross-step preload of the
//     next timestep's first x fragment (x is immutable -> safe across barrier)
//   * one device-wide atomic barrier per timestep
// ---------------------------------------------------------------------------

typedef __bf16 v16bf __attribute__((ext_vector_type(16)));
typedef float  v8f   __attribute__((ext_vector_type(8)));

#define BATCH   64
#define TSTEPS  1024
#define FDIM    256
#define UDIM    512
#define NGATE   2048        // 4*U
#define KTILES  24          // 8 k-tiles from x (F=256) + 16 from h (U=512)
#define NTILES  128         // NGATE/16
#define NBLOCKS 32

#define WKT_STRIDE 131072   // bytes per k-tile in wfrag: 128 ntiles * 32 lanes * 32B
#define OUTROW     ((size_t)TSTEPS * UDIM * 4)   // 2 MB: out stride between batch rows
#define HBUF_BYTES (BATCH * UDIM * 2)            // 64 KB per parity

union Frag32 { uint4 u[2]; v16bf v; };

// ---- prep: zero barrier counters + h ping-pong buffers ----
__global__ void prep_zero(unsigned* __restrict__ p, int n) {
    int i = blockIdx.x * blockDim.x + threadIdx.x;
    for (; i < n; i += gridDim.x * blockDim.x) p[i] = 0u;
}

// ---- prep: fp32 -> bf16 conversion of x ----
__global__ void prep_cvt_bf16(const float* __restrict__ src,
                              __hip_bfloat16* __restrict__ dst, int n) {
    int i = blockIdx.x * blockDim.x + threadIdx.x;
    for (; i < n; i += gridDim.x * blockDim.x) dst[i] = __float2bfloat16(src[i]);
}

// ---- prep: pack [kernel; recurrent_kernel] into WMMA B-fragment layout ----
// wfrag element e = ((kt*128 + ntile)*32 + lane)*16 + j
//   N = ntile*16 + (lane & 15)
//   K = kt*32 + (j>>3)*16 + (lane >> 4)*8 + (j&7)        (B layout: N = L%16, K = (j>>3)*16 + (L>=16)*8 + (j&7))
__global__ void prep_wfrag(const float* __restrict__ kern,
                           const float* __restrict__ rk,
                           __hip_bfloat16* __restrict__ wf) {
    const int total = KTILES * NTILES * 32 * 16;   // 1,572,864
    int e = blockIdx.x * blockDim.x + threadIdx.x;
    for (; e < total; e += gridDim.x * blockDim.x) {
        int j  = e & 15;
        int L  = (e >> 4) & 31;
        int nt = (e >> 9) & 127;
        int kt = e >> 16;
        int N = nt * 16 + (L & 15);
        int K = kt * 32 + (j >> 3) * 16 + (L >> 4) * 8 + (j & 7);
        float v = (K < FDIM) ? kern[(size_t)K * NGATE + N]
                             : rk[(size_t)(K - FDIM) * NGATE + N];
        wf[e] = __float2bfloat16(v);
    }
}

// ---- device-wide split barrier over one per-step counter ----
__device__ __forceinline__ void grid_barrier(unsigned* __restrict__ ctr) {
    __threadfence();                 // release my h-writes to agent scope
    __syncthreads();
    if (threadIdx.x == 0) {
        __hip_atomic_fetch_add(ctr, 1u, __ATOMIC_ACQ_REL, __HIP_MEMORY_SCOPE_AGENT);
        while (__hip_atomic_load(ctr, __ATOMIC_ACQUIRE, __HIP_MEMORY_SCOPE_AGENT)
               < (unsigned)NBLOCKS) {
            __builtin_amdgcn_s_sleep(2);
        }
    }
    __syncthreads();
    __threadfence();                 // acquire other WGs' h-writes
}

// ---------------------------------------------------------------------------
// Persistent scan kernel: 32 WGs x 128 threads (4 waves).
//   blockIdx.x = u-tile (0..31)   -> 16 columns of U
//   wave id    = m-tile (0..3)    -> 16 rows of B
// ---------------------------------------------------------------------------
__global__ __launch_bounds__(128) void lstm_scan(
    const float* __restrict__ bias,            // [2048] fp32
    const __hip_bfloat16* __restrict__ xb,     // [B][T][F] bf16
    const __hip_bfloat16* __restrict__ wfrag,  // [24][128][32][16] bf16
    __hip_bfloat16* __restrict__ hbuf,         // [2][B][U] bf16 ping-pong
    unsigned* __restrict__ counters,           // [T] barrier counters (zeroed)
    float* __restrict__ out)                   // [B][T][U] fp32
{
    const int lane  = threadIdx.x & 31;
    const int m     = threadIdx.x >> 5;     // m-tile 0..3
    const int ut    = blockIdx.x;           // u-tile 0..31
    const int nloc  = lane & 15;
    const int base8 = (lane >> 4) << 3;     // 0 or 8 (upper half-wave)

    // per-lane bias for the 4 gates (gate order: i, f, c, o)
    float bvals[4];
#pragma unroll
    for (int g = 0; g < 4; ++g) bvals[g] = bias[g * UDIM + ut * 16 + nloc];

    const int brow = m * 16 + nloc;         // A-matrix row (M = lane%16)
    const int ucol = ut * 16 + nloc;

    float cstate[8];
#pragma unroll
    for (int r = 0; r < 8; ++r) cstate[r] = 0.0f;

    // ---- hoisted per-lane base pointers ----
    const char* xp = (const char*)xb + ((size_t)brow * TSTEPS * FDIM + base8) * 2;
    const char* h0 = (const char*)hbuf + ((size_t)brow * UDIM + base8) * 2;
    const char* h1 = h0 + HBUF_BYTES;
    char* hs0 = (char*)hbuf + ((size_t)(m * 16 + base8) * UDIM + ucol) * 2;
    char* hs1 = hs0 + HBUF_BYTES;
    char* op0 = (char*)out + ((size_t)(m * 16 + base8) * TSTEPS * UDIM + ucol) * 4;
    char* op4 = op0 + 4 * OUTROW;

    // ---- preload ALL weight fragments into registers: w[4][24] = 768 VGPRs ----
    Frag32 w[4][KTILES];
#pragma unroll
    for (int g = 0; g < 4; ++g) {
        const char* wgb = (const char*)wfrag + (size_t)((g * 32 + ut) * 32 + lane) * 32;
#pragma unroll
        for (int kt = 0; kt < KTILES; ++kt) {
            w[g][kt].u[0] = *(const uint4*)(wgb + (size_t)kt * WKT_STRIDE);
            w[g][kt].u[1] = *(const uint4*)(wgb + (size_t)kt * WKT_STRIDE + 16);
        }
    }

    // A-fragment double buffer; preload k-tile 0 of t=0 (from x)
    Frag32 a[2];
    a[0].u[0] = *(const uint4*)(xp);
    a[0].u[1] = *(const uint4*)(xp + 32);

    for (int t = 0; t < TSTEPS; ++t) {
        const bool odd = (t & 1) != 0;
        const char* hsrc = odd ? h1 : h0;    // read h_{t-1}
        char*       hdst = odd ? hs0 : hs1;  // write h_t (opposite parity)
        // next step's first x fragment (clamped on the final step)
        const char* xnext = (t + 1 < TSTEPS) ? (xp + FDIM * 2) : xp;

        // bias folded into accumulator init
        v8f acc[4];
#pragma unroll
        for (int g = 0; g < 4; ++g)
#pragma unroll
            for (int r = 0; r < 8; ++r) acc[g][r] = bvals[g];

        // ---- software-pipelined K loop: kt 0..7 from x, kt 8..23 from h ----
#pragma unroll
        for (int kt = 0; kt < KTILES; ++kt) {
            // issue next A-fragment loads while current WMMAs execute
            Frag32& anext = a[(kt + 1) & 1];
            if (kt + 1 < 8) {
                anext.u[0] = *(const uint4*)(xp + (kt + 1) * 64);
                anext.u[1] = *(const uint4*)(xp + (kt + 1) * 64 + 32);
            } else if (kt + 1 < KTILES) {
                anext.u[0] = *(const uint4*)(hsrc + (kt + 1 - 8) * 64);
                anext.u[1] = *(const uint4*)(hsrc + (kt + 1 - 8) * 64 + 32);
            } else {
                anext.u[0] = *(const uint4*)(xnext);
                anext.u[1] = *(const uint4*)(xnext + 32);
            }
            const v16bf av = a[kt & 1].v;
#pragma unroll
            for (int g = 0; g < 4; ++g) {
                acc[g] = __builtin_amdgcn_wmma_f32_16x16x32_bf16(
                    false, av, false, w[g][kt].v, (short)0, acc[g], false, false);
            }
        }
        // after the loop, a[0] holds k-tile 0 of timestep t+1

        // gates + cell update + outputs (C/D layout: M = r + (lane>=16)*8, N = lane%16)
#pragma unroll
        for (int r = 0; r < 8; ++r) {
            float zi = acc[0][r];
            float zf = acc[1][r];
            float zg = acc[2][r];              // linear candidate
            float zo = acc[3][r];
            float gi = 1.0f / (1.0f + __expf(-zi));
            float gf = 1.0f / (1.0f + __expf(-zf));
            float go = 1.0f / (1.0f + __expf(-zo));
            float c  = gf * cstate[r] + gi * zg;
            cstate[r] = c;
            float h  = go * c;                 // linear output activation
            *(__hip_bfloat16*)(hdst + (size_t)r * (UDIM * 2)) = __float2bfloat16(h);
            // external tanh: 1 - 2/(e^{2h}+1), exact at +/-inf
            char* obase = (r < 4) ? op0 : op4;
            *(float*)(obase + (size_t)(r & 3) * OUTROW) =
                1.0f - 2.0f / (__expf(2.0f * h) + 1.0f);
        }

        xp  += FDIM * 2;      // next timestep's x slice
        op0 += UDIM * 4;      // next timestep's output column
        op4 += UDIM * 4;

        grid_barrier(&counters[t]);
    }
}

// ---------------------------------------------------------------------------
// Workspace layout (bytes):
//   [0,            3,145,728)  wfrag   (24*128*32*16 bf16)
//   [3,145,728,    3,276,800)  hbuf    (2*64*512 bf16)
//   [3,276,800,    3,280,896)  counters(1024 u32)
//   [3,280,896,   36,835,328)  xb      (64*1024*256 bf16)
// ---------------------------------------------------------------------------
extern "C" void kernel_launch(void* const* d_in, const int* in_sizes, int n_in,
                              void* d_out, int out_size, void* d_ws, size_t ws_size,
                              hipStream_t stream) {
    const float* x    = (const float*)d_in[0];   // [64,1024,256]
    const float* kern = (const float*)d_in[1];   // [256,2048]
    const float* rk   = (const float*)d_in[2];   // [512,2048]
    const float* bias = (const float*)d_in[3];   // [2048]
    float* out = (float*)d_out;

    char* ws = (char*)d_ws;
    __hip_bfloat16* wfrag    = (__hip_bfloat16*)(ws);
    __hip_bfloat16* hbuf     = (__hip_bfloat16*)(ws + 3145728);
    unsigned*       counters = (unsigned*)(ws + 3145728 + 131072);
    __hip_bfloat16* xb       = (__hip_bfloat16*)(ws + 3145728 + 131072 + 4096);

    // zero hbuf (both parities) + counters: adjacent region, 33792 u32
    prep_zero<<<132, 256, 0, stream>>>((unsigned*)hbuf, (131072 + 4096) / 4);
    // x -> bf16
    prep_cvt_bf16<<<4096, 256, 0, stream>>>(x, xb, BATCH * TSTEPS * FDIM);
    // pack weights into B-fragment layout
    prep_wfrag<<<3072, 256, 0, stream>>>(kern, rk, wfrag);
    // persistent scan
    lstm_scan<<<NBLOCKS, 128, 0, stream>>>(bias, xb, wfrag, hbuf, counters, out);
}
